// Transformer_38869454029043
// MI455X (gfx1250) — matrix operand order repaired and checked by reading, not compiled, not run on hardware
//
#include <hip/hip_runtime.h>
#include <hip/hip_bf16.h>
#include <math.h>

// ---------------- problem constants ----------------
#define Bn  16
#define Sn  512
#define En  128
#define Hn  768
#define NHn 12
#define DWn 64
#define FFn 3072
#define NLn 12
#define MROWS (Bn*Sn)           // 8192 token rows
#define EPSf 1e-12f

typedef _Float16 half_t;
typedef __attribute__((ext_vector_type(16))) _Float16 v16h;
typedef __attribute__((ext_vector_type(8)))  float    v8f;

union Frag16 { uint4 u[2]; v16h h; };

// strided gather -> fragment (B-layout: lane = N column, K[kof..kof+7],[16+kof..23+kof])
static __device__ inline v16h ld_frag16(const half_t* p, int ld, int kof) {
    v16h f;
#pragma unroll
    for (int i = 0; i < 8; ++i) {
        f[i] = p[(kof + i) * ld];
        f[8 + i] = p[(16 + kof + i) * ld];
    }
    return f;
}
// A-matrix 16x32 f16 layout (ISA 7.12.2): lane<16 -> K[0..7],[16..23]; lane>=16 -> K[8..15],[24..31]
static __device__ inline v16h ld_fragA(const half_t* rowp, int kof) {
    Frag16 f;
    f.u[0] = *reinterpret_cast<const uint4*>(rowp + kof);
    f.u[1] = *reinterpret_cast<const uint4*>(rowp + kof + 16);
    return f.h;
}

// CDNA5 async global -> LDS copy, 16 bytes per lane, tracked by ASYNCcnt.
static __device__ inline void async_b128(half_t* lds_ptr, const half_t* gptr) {
    unsigned lds_addr = (unsigned)(uintptr_t)lds_ptr;  // low 32 bits = LDS byte offset
    unsigned long long ga = (unsigned long long)(uintptr_t)gptr;
    asm volatile("global_load_async_to_lds_b128 %0, %1, off"
                 :: "v"(lds_addr), "v"(ga) : "memory");
}
// copy 16 halfs (32B) via two async b128
static __device__ inline void async_16h(half_t* lds_ptr, const half_t* gptr) {
    async_b128(lds_ptr, gptr);
    async_b128(lds_ptr + 8, gptr + 8);
}

static __device__ inline float gelu_f(float x) {
    return 0.5f * x * (1.0f + erff(x * 0.70710678118654752f));
}
static __device__ inline float hmax16(float v) {
    v = fmaxf(v, __shfl_xor(v, 1, 32));
    v = fmaxf(v, __shfl_xor(v, 2, 32));
    v = fmaxf(v, __shfl_xor(v, 4, 32));
    v = fmaxf(v, __shfl_xor(v, 8, 32));
    return v;
}
static __device__ inline float hsum16(float v) {
    v += __shfl_xor(v, 1, 32);
    v += __shfl_xor(v, 2, 32);
    v += __shfl_xor(v, 4, 32);
    v += __shfl_xor(v, 8, 32);
    return v;
}

// ---------------- f32 -> f16 weight convert ----------------
__global__ void cvt_f16_kernel(const float* __restrict__ src, half_t* __restrict__ dst, int n) {
    int i = blockIdx.x * blockDim.x + threadIdx.x;
    if (i < n) dst[i] = (half_t)src[i];
}

// ---------------- embedding + LN ----------------
__global__ __launch_bounds__(256)
void embed_ln_kernel(const int* __restrict__ x, const float* __restrict__ tok,
                     const float* __restrict__ w2, const float* __restrict__ b2,
                     const float* __restrict__ pos, const float* __restrict__ g,
                     const float* __restrict__ be, float* __restrict__ hF,
                     half_t* __restrict__ hH) {
    __shared__ float emb[En];
    __shared__ float vals[Hn];
    __shared__ float red[256], red2[256];
    const int row = blockIdx.x, t = threadIdx.x;
    const int tokid = x[row];
    if (t < En) emb[t] = tok[(size_t)tokid * En + t];
    __syncthreads();
    const int s = row % Sn;
    float lsum = 0.f, lsq = 0.f;
#pragma unroll
    for (int i = 0; i < 3; ++i) {
        const int c = t + i * 256;
        float acc = b2[c] + pos[(size_t)s * Hn + c];
        for (int k = 0; k < En; ++k) acc += emb[k] * w2[(size_t)k * Hn + c];
        vals[c] = acc; lsum += acc; lsq += acc * acc;
    }
    red[t] = lsum; red2[t] = lsq;
    __syncthreads();
    for (int st = 128; st > 0; st >>= 1) {
        if (t < st) { red[t] += red[t + st]; red2[t] += red2[t + st]; }
        __syncthreads();
    }
    const float mean = red[0] * (1.0f / Hn);
    const float var  = red2[0] * (1.0f / Hn) - mean * mean;
    const float rstd = rsqrtf(var + EPSf);
#pragma unroll
    for (int i = 0; i < 3; ++i) {
        const int c = t + i * 256;
        const float y = g[c] * (vals[c] - mean) * rstd + be[c];
        const size_t idx = (size_t)row * Hn + c;
        hF[idx] = y; hH[idx] = (half_t)y;
    }
}

// ---------------- residual + LN ----------------
__global__ __launch_bounds__(256)
void residual_ln_kernel(const float* __restrict__ a, const float* __restrict__ r,
                        const float* __restrict__ g, const float* __restrict__ be,
                        float* __restrict__ hF, half_t* __restrict__ hH) {
    __shared__ float vals[Hn];
    __shared__ float red[256], red2[256];
    const int row = blockIdx.x, t = threadIdx.x;
    float lsum = 0.f, lsq = 0.f;
#pragma unroll
    for (int i = 0; i < 3; ++i) {
        const int c = t + i * 256;
        const size_t idx = (size_t)row * Hn + c;
        const float v = a[idx] + r[idx];
        vals[c] = v; lsum += v; lsq += v * v;
    }
    red[t] = lsum; red2[t] = lsq;
    __syncthreads();
    for (int st = 128; st > 0; st >>= 1) {
        if (t < st) { red[t] += red[t + st]; red2[t] += red2[t + st]; }
        __syncthreads();
    }
    const float mean = red[0] * (1.0f / Hn);
    const float var  = red2[0] * (1.0f / Hn) - mean * mean;
    const float rstd = rsqrtf(var + EPSf);
#pragma unroll
    for (int i = 0; i < 3; ++i) {
        const int c = t + i * 256;
        const float y = g[c] * (vals[c] - mean) * rstd + be[c];
        const size_t idx = (size_t)row * Hn + c;
        hF[idx] = y; hH[idx] = (half_t)y;
    }
}

// ---------------- WMMA f16 GEMM:  C[M,N] = A[M,K] * W[K,N] + bias ----------------
// block tile 128x128, 8 waves (4x2), wave tile 32x64 = 2x4 WMMA subtiles, K step 32.
// Double-buffered LDS fed by CDNA5 async global->LDS loads (ASYNCcnt).
#define GLDA 40    // 32 + 8 pad (halfs)
#define GLDB 136   // 128 + 8 pad (halfs)
#define ABUF (128 * GLDA)
#define BBUF (32 * GLDB)
__global__ __launch_bounds__(256)
void gemm_f16_kernel(const half_t* __restrict__ A, const half_t* __restrict__ W,
                     const float* __restrict__ bias, float* __restrict__ outF,
                     half_t* __restrict__ outH, int N, int K, int act) {
    __shared__ half_t ldsA[2 * ABUF];
    __shared__ half_t ldsB[2 * BBUF];
    const int t = threadIdx.x, lane = t & 31, wave = t >> 5;
    const int wm = wave & 3, wn = wave >> 2;
    const int row0 = blockIdx.x * 128, n0 = blockIdx.y * 128;
    const int kof = (lane < 16) ? 0 : 8;

    // per-thread staging coordinates (4 async b128 per thread per stage)
    const int ar = t >> 1, akc = (t & 1) * 16;          // A: 128 rows x 32 halfs
    const int bkr = t >> 3, bnc = (t & 7) * 16;         // B: 32 rows x 128 halfs
    const half_t* aSrcBase = A + (size_t)(row0 + ar) * K + akc;
    const half_t* bSrcBase = W + (size_t)bkr * N + n0 + bnc;

    // prologue: stage k-step 0 into buffer 0
    async_16h(&ldsA[ar * GLDA + akc], aSrcBase);
    async_16h(&ldsB[bkr * GLDB + bnc], bSrcBase);

    v8f acc[2][4] = {};
    const int ksteps = K >> 5;
    for (int ks = 0; ks < ksteps; ++ks) {
        const int p = ks & 1;
        if (ks + 1 < ksteps) {  // stage next k-step into the other buffer
            const int k1 = (ks + 1) << 5;
            async_16h(&ldsA[(p ^ 1) * ABUF + ar * GLDA + akc], aSrcBase + k1);
            async_16h(&ldsB[(p ^ 1) * BBUF + bkr * GLDB + bnc], bSrcBase + (size_t)k1 * N);
            asm volatile("s_wait_asynccnt 0x4" ::: "memory");  // current stage landed
        } else {
            asm volatile("s_wait_asynccnt 0x0" ::: "memory");
        }
        __syncthreads();

        const half_t* bufA = &ldsA[p * ABUF];
        const half_t* bufB = &ldsB[p * BBUF];
        v16h af[2], bf[4];
#pragma unroll
        for (int mi = 0; mi < 2; ++mi)
            af[mi] = ld_fragA(&bufA[(wm * 32 + mi * 16 + (lane & 15)) * GLDA], kof);
#pragma unroll
        for (int ni = 0; ni < 4; ++ni)
            bf[ni] = ld_frag16(&bufB[(lane & 15) + wn * 64 + ni * 16], GLDB, kof);
#pragma unroll
        for (int mi = 0; mi < 2; ++mi)
#pragma unroll
            for (int ni = 0; ni < 4; ++ni)
                acc[mi][ni] = __builtin_amdgcn_wmma_f32_16x16x32_f16(
                    false, af[mi], false, bf[ni], (short)0, acc[mi][ni], false, false);
        __syncthreads();
    }
    // epilogue (C layout: VGPR j -> row j + 8*(lane>=16), col lane%16)
#pragma unroll
    for (int mi = 0; mi < 2; ++mi)
#pragma unroll
        for (int ni = 0; ni < 4; ++ni) {
            const int gn = n0 + wn * 64 + ni * 16 + (lane & 15);
            const float bv = bias[gn];
#pragma unroll
            for (int j = 0; j < 8; ++j) {
                const int gm = row0 + wm * 32 + mi * 16 + j + ((lane >> 4) << 3);
                float v = acc[mi][ni][j] + bv;
                if (act == 1) v = gelu_f(v);
                const size_t idx = (size_t)gm * N + gn;
                if (outF) outF[idx] = v;
                if (outH) outH[idx] = (half_t)v;
            }
        }
}

// ---------------- flash attention ----------------
// grid (S/64, NH, B), 128 threads = 4 waves; each wave owns a 16-row Q stripe.
#define LDKT 40  // 32 kv cols + 8 pad
#define LDV  72  // 64 dims + 8 pad
__global__ __launch_bounds__(128)
void attn_kernel(const half_t* __restrict__ qkv, const float* __restrict__ mask,
                 float* __restrict__ oF, half_t* __restrict__ oH) {
    __shared__ half_t ldsKT[64 * LDKT];      // K^T: [dim][kv]
    __shared__ half_t ldsV[32 * LDV];        // V:   [kv][dim]
    __shared__ half_t ldsP[4 * 16 * 32];     // per-wave P tile
    const int t = threadIdx.x, lane = t & 31, wave = t >> 5;
    const int head = blockIdx.y, b = blockIdx.z;
    const int q0 = blockIdx.x * 64 + wave * 16;
    const size_t ld = 3 * Hn;
    const int kof = (lane < 16) ? 0 : 8;

    // persistent Q fragments: 16 rows x 64 dims = 2 A-operands
    const half_t* qp = qkv + (size_t)(b * Sn + q0 + (lane & 15)) * ld + head * DWn;
    v16h qf[2];
#pragma unroll
    for (int kk = 0; kk < 2; ++kk) qf[kk] = ld_fragA(qp + kk * 32, kof);

    float m8[8], l8[8];
    v8f o4[4] = {};
#pragma unroll
    for (int j = 0; j < 8; ++j) { m8[j] = -1e30f; l8[j] = 0.f; }
    const float* mrow = mask + (size_t)b * Sn;
    const float scale = 0.125f;  // 1/sqrt(64)
    half_t* pbase = &ldsP[wave * 512];
    const int kv = t >> 2, d0 = (t & 3) * 16;

    for (int kc = 0; kc < Sn / 32; ++kc) {
        {   // cooperative stage: V via async global->LDS, K via VGPR transpose
            const half_t* kp = qkv + (size_t)(b * Sn + kc * 32 + kv) * ld + Hn + head * DWn + d0;
            async_16h(&ldsV[kv * LDV + d0], kp + Hn);  // V lives at +2H
            Frag16 kf;
            kf.u[0] = *reinterpret_cast<const uint4*>(kp);
            kf.u[1] = *reinterpret_cast<const uint4*>(kp + 8);
            half_t tmp[16];
            *reinterpret_cast<uint4*>(&tmp[0]) = kf.u[0];
            *reinterpret_cast<uint4*>(&tmp[8]) = kf.u[1];
#pragma unroll
            for (int i = 0; i < 16; ++i) ldsKT[(d0 + i) * LDKT + kv] = tmp[i];
        }
        asm volatile("s_wait_asynccnt 0x0" ::: "memory");
        __syncthreads();

        // S = Q K^T  (16 x 32 scores as two 16x16 tiles)
        v8f s0 = {}, s1 = {};
#pragma unroll
        for (int kk = 0; kk < 2; ++kk) {
            v16h b0 = ld_frag16(&ldsKT[(kk * 32) * LDKT + 0 + (lane & 15)], LDKT, kof);
            v16h b1 = ld_frag16(&ldsKT[(kk * 32) * LDKT + 16 + (lane & 15)], LDKT, kof);
            s0 = __builtin_amdgcn_wmma_f32_16x16x32_f16(false, qf[kk], false, b0, (short)0, s0, false, false);
            s1 = __builtin_amdgcn_wmma_f32_16x16x32_f16(false, qf[kk], false, b1, (short)0, s1, false, false);
        }
        const float am0 = -10000.f * (1.f - mrow[kc * 32 + (lane & 15)]);
        const float am1 = -10000.f * (1.f - mrow[kc * 32 + 16 + (lane & 15)]);

        // online softmax per row (rows j+8*(lane>=16); 16 lanes per row half)
#pragma unroll
        for (int j = 0; j < 8; ++j) {
            const float a0 = s0[j] * scale + am0;
            const float a1 = s1[j] * scale + am1;
            const float rm = hmax16(fmaxf(a0, a1));
            const float mn = fmaxf(m8[j], rm);
            const float sc = __expf(m8[j] - mn);
            m8[j] = mn;
            const float p0 = __expf(a0 - mn);
            const float p1 = __expf(a1 - mn);
            l8[j] = l8[j] * sc + hsum16(p0 + p1);
#pragma unroll
            for (int tt = 0; tt < 4; ++tt) o4[tt][j] *= sc;
            const int pr = j + ((lane >> 4) << 3);
            pbase[pr * 32 + (lane & 15)]      = (half_t)p0;
            pbase[pr * 32 + 16 + (lane & 15)] = (half_t)p1;
        }
        asm volatile("s_wait_dscnt 0x0" ::: "memory");  // in-wave P store->load

        // O += P (16x32) @ V (32x64)
        const v16h pf = ld_fragA(pbase + (lane & 15) * 32, kof);
#pragma unroll
        for (int tt = 0; tt < 4; ++tt) {
            v16h vb = ld_frag16(&ldsV[tt * 16 + (lane & 15)], LDV, kof);
            o4[tt] = __builtin_amdgcn_wmma_f32_16x16x32_f16(false, pf, false, vb, (short)0, o4[tt], false, false);
        }
        __syncthreads();
    }

    // normalize and write
#pragma unroll
    for (int j = 0; j < 8; ++j) {
        const float inv = 1.f / l8[j];
        const int srow = q0 + j + ((lane >> 4) << 3);
        const size_t base = (size_t)(b * Sn + srow) * Hn + head * DWn;
#pragma unroll
        for (int tt = 0; tt < 4; ++tt) {
            const float v = o4[tt][j] * inv;
            const size_t idx = base + tt * 16 + (lane & 15);
            oF[idx] = v;
            oH[idx] = (half_t)v;
        }
    }
}

// ---------------- host side ----------------
extern "C" void kernel_launch(void* const* d_in, const int* in_sizes, int n_in,
                              void* d_out, int out_size, void* d_ws, size_t ws_size,
                              hipStream_t stream) {
    (void)in_sizes; (void)n_in; (void)out_size; (void)ws_size;
    const int*   x      = (const int*)d_in[0];
    // d_in[1] = seg (unused by reference forward)
    const float* mask   = (const float*)d_in[2];
    const float* tok    = (const float*)d_in[3];
    const float* W_tok2 = (const float*)d_in[4];
    const float* b_tok2 = (const float*)d_in[5];
    const float* pos    = (const float*)d_in[6];
    const float* g_e    = (const float*)d_in[7];
    const float* b_e    = (const float*)d_in[8];
    const float* Wqkv   = (const float*)d_in[9];
    const float* bqkv   = (const float*)d_in[10];
    const float* Wproj  = (const float*)d_in[11];
    const float* bproj  = (const float*)d_in[12];
    const float* g1     = (const float*)d_in[13];
    const float* b1     = (const float*)d_in[14];
    const float* Wff1   = (const float*)d_in[15];
    const float* bff1   = (const float*)d_in[16];
    const float* Wff2   = (const float*)d_in[17];
    const float* bff2   = (const float*)d_in[18];
    const float* g2     = (const float*)d_in[19];
    const float* b2     = (const float*)d_in[20];

    // bump-allocate workspace (256B aligned)
    char* ws = (char*)d_ws;
    size_t off = 0;
    auto alloc = [&](size_t bytes) -> void* {
        void* p = ws + off;
        off = (off + bytes + 255) & ~(size_t)255;
        return p;
    };
    half_t* wqkv_h  = (half_t*)alloc((size_t)Hn * 3 * Hn * sizeof(half_t));
    half_t* wproj_h = (half_t*)alloc((size_t)Hn * Hn * sizeof(half_t));
    half_t* wff1_h  = (half_t*)alloc((size_t)Hn * FFn * sizeof(half_t));
    half_t* wff2_h  = (half_t*)alloc((size_t)FFn * Hn * sizeof(half_t));
    float*  h_f     = (float*) alloc((size_t)MROWS * Hn * sizeof(float));
    half_t* h_h     = (half_t*)alloc((size_t)MROWS * Hn * sizeof(half_t));
    float*  o_f     = (float*) alloc((size_t)MROWS * Hn * sizeof(float));
    half_t* o_h     = (half_t*)alloc((size_t)MROWS * Hn * sizeof(half_t));
    float*  tmp_f   = (float*) alloc((size_t)MROWS * Hn * sizeof(float));
    // qkv (M x 3H halfs) and ff activations (M x FF halfs) have disjoint lifetimes -> alias
    half_t* scratch = (half_t*)alloc((size_t)MROWS * FFn * sizeof(half_t));
    half_t* qkv_h   = scratch;
    half_t* ff_h    = scratch;

    // 1) convert weights to f16
    {
        int n;
        n = Hn * 3 * Hn; cvt_f16_kernel<<<(n + 255) / 256, 256, 0, stream>>>(Wqkv,  wqkv_h,  n);
        n = Hn * Hn;     cvt_f16_kernel<<<(n + 255) / 256, 256, 0, stream>>>(Wproj, wproj_h, n);
        n = Hn * FFn;    cvt_f16_kernel<<<(n + 255) / 256, 256, 0, stream>>>(Wff1,  wff1_h,  n);
        n = FFn * Hn;    cvt_f16_kernel<<<(n + 255) / 256, 256, 0, stream>>>(Wff2,  wff2_h,  n);
    }
    // 2) embedding + LN
    embed_ln_kernel<<<MROWS, 256, 0, stream>>>(x, tok, W_tok2, b_tok2, pos, g_e, b_e, h_f, h_h);

    // 3) 12 layers, shared weights
    const dim3 gQKV (MROWS / 128, (3 * Hn) / 128);
    const dim3 gPROJ(MROWS / 128, Hn / 128);
    const dim3 gFF1 (MROWS / 128, FFn / 128);
    const dim3 gFF2 (MROWS / 128, Hn / 128);
    const dim3 gATT (Sn / 64, NHn, Bn);
    for (int layer = 0; layer < NLn; ++layer) {
        gemm_f16_kernel<<<gQKV, 256, 0, stream>>>(h_h, wqkv_h, bqkv, nullptr, qkv_h, 3 * Hn, Hn, 0);
        attn_kernel<<<gATT, 128, 0, stream>>>(qkv_h, mask, o_f, o_h);
        gemm_f16_kernel<<<gPROJ, 256, 0, stream>>>(o_h, wproj_h, bproj, tmp_f, nullptr, Hn, Hn, 0);
        residual_ln_kernel<<<MROWS, 256, 0, stream>>>(o_f, tmp_f, g1, b1, h_f, h_h);
        gemm_f16_kernel<<<gFF1, 256, 0, stream>>>(h_h, wff1_h, bff1, nullptr, ff_h, FFn, Hn, 1);
        gemm_f16_kernel<<<gFF2, 256, 0, stream>>>(ff_h, wff2_h, bff2, tmp_f, nullptr, Hn, FFn, 0);
        residual_ln_kernel<<<MROWS, 256, 0, stream>>>(h_f, tmp_f, g2, b2, h_f, h_h);
    }
    // 4) output
    hipMemcpyAsync(d_out, h_f, (size_t)MROWS * Hn * sizeof(float),
                   hipMemcpyDeviceToDevice, stream);
}
